// SelfAttentionLayer_56006373539863
// MI455X (gfx1250) — matrix operand order repaired and checked by reading, not compiled, not run on hardware
//
#include <hip/hip_runtime.h>

// ---------------------------------------------------------------------------
// Self-attention layer for MI455X (gfx1250): f16 WMMA everywhere, f32 accum.
// Software-pipelined GEMM fragment loads + double-buffered async-to-LDS
// (ASYNCcnt) staging in flash attention.
// ---------------------------------------------------------------------------

typedef _Float16 h8_t  __attribute__((ext_vector_type(8)));
typedef _Float16 h16_t __attribute__((ext_vector_type(16)));
typedef float    v8f   __attribute__((ext_vector_type(8)));
typedef int      i4_t  __attribute__((ext_vector_type(4)));

#define D_MODEL  512
#define N_HEAD   8
#define HEAD_DIM 64
#define BATCH    8
#define SEQ      2048
#define NROWS    (BATCH * SEQ)   // 16384

// ---- gfx1250 async global->LDS copy (ASYNCcnt) ----------------------------
#if defined(__has_builtin)
#if __has_builtin(__builtin_amdgcn_global_load_async_to_lds_b128)
#define HAVE_ASYNC_LDS 1
#endif
#endif

#if defined(HAVE_ASYNC_LDS)
__device__ __forceinline__ void async_cp16(const _Float16* g, _Float16* l) {
  // Builtin signature: (v4i addrspace(1)*, v4i addrspace(3)*, imm offset, imm cpol)
  __builtin_amdgcn_global_load_async_to_lds_b128(
      (__attribute__((address_space(1))) i4_t*)(g),
      (__attribute__((address_space(3))) i4_t*)(l), 0, 0);
}
#if __has_builtin(__builtin_amdgcn_s_wait_asynccnt)
#define WAIT_ASYNC(n) __builtin_amdgcn_s_wait_asynccnt(n)
#else
#define WAIT_ASYNC(n) asm volatile("s_wait_asynccnt " #n ::: "memory")
#endif
#else
#define WAIT_ASYNC(n) ((void)0)
#endif

__device__ __forceinline__ v8f wmma16x16x32(h16_t a, h16_t b, v8f c) {
  // D = A(16x32 f16) x B(32x16 f16) + C(16x16 f32)
  return __builtin_amdgcn_wmma_f32_16x16x32_f16(
      /*neg_a=*/false, a, /*neg_b=*/false, b,
      /*c_mod=*/(short)0, c, /*reuse_a=*/false, /*reuse_b=*/false);
}

// A fragment (16x32, 16-bit): lane (l&15) supplies row M; hi = l>>4.
// elements 0..7 -> K = hi*8 + 0..7 ; elements 8..15 -> K = 16 + hi*8 + 0..7
__device__ __forceinline__ h16_t load_a_frag(const _Float16* rowk, int hi) {
  union { h16_t v; h8_t h[2]; } u;
  u.h[0] = *(const h8_t*)(rowk + hi * 8);
  u.h[1] = *(const h8_t*)(rowk + 16 + hi * 8);
  return u.v;
}

// B fragment (32x16, 16-bit): lane (l&15) supplies column N;
// elements 0..7 -> K = hi*8 + 0..7 ; elements 8..15 -> K = 16 + hi*8 + 0..7
__device__ __forceinline__ h16_t load_b_frag(const _Float16* colk, int hi) {
  union { h16_t v; h8_t h[2]; } u;
  u.h[0] = *(const h8_t*)(colk + hi * 8);
  u.h[1] = *(const h8_t*)(colk + 16 + hi * 8);
  return u.v;
}

// ---------------------------------------------------------------------------
// f32 -> f16 conversion
// ---------------------------------------------------------------------------
__global__ void cvt_f32_f16(const float* __restrict__ src,
                            _Float16* __restrict__ dst, int n) {
  int i = blockIdx.x * blockDim.x + threadIdx.x;
  if (i < n) dst[i] = (_Float16)src[i];
}

// ---------------------------------------------------------------------------
// QKV projection: y = x @ W^T + b, software-pipelined WMMA GEMM.
// Q,K outputs: [B,H,S,64] f16.  V output: [B,H,64,S] f16 (pre-transposed so
// the attention kernel can async-copy V^T tiles with contiguous b128 ops).
// grid: (NROWS/64, D_MODEL/128, 3)   block: 256 (8 waves)
// ---------------------------------------------------------------------------
__global__ void qkv_proj_kernel(const _Float16* __restrict__ xh,
                                const _Float16* __restrict__ whAll,
                                const float* __restrict__ bq,
                                const float* __restrict__ bk,
                                const float* __restrict__ bv,
                                _Float16* __restrict__ qh,
                                _Float16* __restrict__ kh,
                                _Float16* __restrict__ vh) {
  const int z = blockIdx.z;
  const _Float16* wh = whAll + (size_t)z * D_MODEL * D_MODEL;
  const float* bias = (z == 0) ? bq : (z == 1) ? bk : bv;
  _Float16* dst = (z == 0) ? qh : (z == 1) ? kh : vh;

  const int w    = threadIdx.x >> 5;
  const int lane = threadIdx.x & 31;
  const int ln   = lane & 15, hi = lane >> 4;
  const int wm = w & 3, wn = w >> 2;
  const int mBase = blockIdx.x * 64 + wm * 16;
  const int nBase = blockIdx.y * 128 + wn * 64;

  const _Float16* arow = xh + (size_t)(mBase + ln) * D_MODEL;
  const _Float16* brow = wh + (size_t)(nBase + ln) * D_MODEL;
  __builtin_prefetch(brow, 0, 0);

  v8f acc[4] = {};
  // Pipelined fragment loads: prefetch step k+1 while WMMA consumes step k.
  h16_t a_cur = load_a_frag(arow, hi);
  h16_t b_cur[4];
#pragma unroll
  for (int nt = 0; nt < 4; ++nt)
    b_cur[nt] = load_b_frag(brow + (size_t)nt * 16 * D_MODEL, hi);

  for (int kb = 0; kb < D_MODEL - 32; kb += 32) {
    h16_t a_nxt = load_a_frag(arow + kb + 32, hi);
    h16_t b_nxt[4];
#pragma unroll
    for (int nt = 0; nt < 4; ++nt)
      b_nxt[nt] = load_b_frag(brow + (size_t)nt * 16 * D_MODEL + kb + 32, hi);
#pragma unroll
    for (int nt = 0; nt < 4; ++nt)
      acc[nt] = wmma16x16x32(a_cur, b_cur[nt], acc[nt]);
    a_cur = a_nxt;
#pragma unroll
    for (int nt = 0; nt < 4; ++nt) b_cur[nt] = b_nxt[nt];
  }
#pragma unroll
  for (int nt = 0; nt < 4; ++nt)
    acc[nt] = wmma16x16x32(a_cur, b_cur[nt], acc[nt]);

#pragma unroll
  for (int nt = 0; nt < 4; ++nt) {
#pragma unroll
    for (int r = 0; r < 8; ++r) {
      int row = mBase + hi * 8 + r;        // = b*SEQ + s
      int col = nBase + nt * 16 + ln;      // = h*64 + d
      float val = acc[nt][r] + bias[col];
      int b = row >> 11, s = row & (SEQ - 1);
      int h = col >> 6,  d = col & (HEAD_DIM - 1);
      size_t idx;
      if (z == 2)  // V stored transposed per head: [B,H,64,S]
        idx = ((size_t)(b * N_HEAD + h) * HEAD_DIM + d) * SEQ + s;
      else         // Q,K: [B,H,S,64]
        idx = ((size_t)(b * N_HEAD + h) * SEQ + s) * HEAD_DIM + d;
      dst[idx] = (_Float16)val;
    }
  }
}

// ---------------------------------------------------------------------------
// Tile copy: 64 rows x 64 halves (128B rows), 128 threads, 4 x b128 each.
// ---------------------------------------------------------------------------
__device__ __forceinline__ void tile_copy(const _Float16* grow0, size_t gstride,
                                          _Float16* lds, int tid) {
  const int row = tid >> 1, halfSel = tid & 1;
  const _Float16* src = grow0 + (size_t)row * gstride + halfSel * 32;
  _Float16* dstp = lds + row * 64 + halfSel * 32;
#pragma unroll
  for (int j = 0; j < 4; ++j) {
#if defined(HAVE_ASYNC_LDS)
    async_cp16(src + j * 8, dstp + j * 8);
#else
    *(h8_t*)(dstp + j * 8) = *(const h8_t*)(src + j * 8);
#endif
  }
}

// ---------------------------------------------------------------------------
// Flash attention per (b,h). Block = 128 threads (4 waves); each wave owns a
// 16-row Q tile (Q fragments register-resident). K and V^T tiles are staged
// into double-buffered LDS with async-to-LDS copies: prefetch tile kt+1 into
// the alternate buffer, then s_wait_asynccnt 8 (in-order completion) +
// barrier makes tile kt visible while the prefetch stays in flight.
// grid: (SEQ/64, BATCH*N_HEAD)   block: 128
// ---------------------------------------------------------------------------
__global__ void flash_attn_kernel(const _Float16* __restrict__ qh,
                                  const _Float16* __restrict__ kh,
                                  const _Float16* __restrict__ vt,
                                  _Float16* __restrict__ ctx) {
  __shared__ _Float16 Ksh[2][64 * 64];   // keys row-major [key][d], 2 buffers
  __shared__ _Float16 VTsh[2][64 * 64];  // V^T [d][key], 2 buffers
  __shared__ _Float16 Psh[4 * 16 * 64];  // per-wave P relayout (C->A format)

  const int bh = blockIdx.y;
  const int b = bh >> 3, h = bh & 7;
  const int qBase = blockIdx.x * 64;
  const int tid = threadIdx.x;
  const int w = tid >> 5, lane = tid & 31, ln = lane & 15, hi = lane >> 4;
  const size_t headOff = (size_t)bh * SEQ * HEAD_DIM;  // also size of [64,S]

  const _Float16* kbase  = kh + headOff;               // [S][64]
  const _Float16* vtbase = vt + headOff;               // [64][S]

  // Q fragments: rows qBase + w*16 .. +15, K-dim 64 = two 16x32 fragments.
  const _Float16* qrow = qh + headOff + (size_t)(qBase + w * 16 + ln) * HEAD_DIM;
  h16_t qa0 = load_a_frag(qrow + 0, hi);
  h16_t qa1 = load_a_frag(qrow + 32, hi);

  v8f oa[4] = {};
  float runMax[8], runSum[8];
#pragma unroll
  for (int r = 0; r < 8; ++r) { runMax[r] = -1e30f; runSum[r] = 0.f; }

  const float scale = 0.125f;  // 1/sqrt(HEAD_DIM)
  _Float16* pw = Psh + w * (16 * 64);

  int cur = 0;
  // Prologue: stage tile 0 into buffer 0 (4 K-ops + 4 V-ops per thread).
  tile_copy(kbase, HEAD_DIM, Ksh[0], tid);
  tile_copy(vtbase, SEQ, VTsh[0], tid);

  for (int kt = 0; kt < SEQ / 64; ++kt) {
    if (kt + 1 < SEQ / 64) {
      // Prefetch next tile into the alternate buffer, then wait for the
      // 8 older ops (this tile) to land; prefetch stays in flight.
      tile_copy(kbase + (size_t)(kt + 1) * 64 * HEAD_DIM, HEAD_DIM,
                Ksh[cur ^ 1], tid);
      tile_copy(vtbase + (kt + 1) * 64, SEQ, VTsh[cur ^ 1], tid);
      WAIT_ASYNC(8);
    } else {
      WAIT_ASYNC(0);
    }
    __syncthreads();  // all waves' copies of tile kt visible

    const _Float16* Kc  = Ksh[cur];
    const _Float16* VTc = VTsh[cur];

    // S = (Q K^T) — B-fragment column n is key row n of Kc (contiguous).
    v8f sacc[4] = {};
#pragma unroll
    for (int nt = 0; nt < 4; ++nt) {
      const _Float16* kcol = Kc + (nt * 16 + ln) * 64;
      sacc[nt] = wmma16x16x32(qa0, load_b_frag(kcol, hi), sacc[nt]);
      sacc[nt] = wmma16x16x32(qa1, load_b_frag(kcol + 32, hi), sacc[nt]);
    }

    // Online softmax per row (row = one 16-lane half of the wave32).
#pragma unroll
    for (int r = 0; r < 8; ++r) {
      float lm = -1e30f;
#pragma unroll
      for (int nt = 0; nt < 4; ++nt) lm = fmaxf(lm, sacc[nt][r] * scale);
#pragma unroll
      for (int off = 8; off >= 1; off >>= 1)
        lm = fmaxf(lm, __shfl_xor(lm, off, 16));
      float nm = fmaxf(runMax[r], lm);
      float alpha = __expf(runMax[r] - nm);
      runMax[r] = nm;
      float ls = 0.f;
#pragma unroll
      for (int nt = 0; nt < 4; ++nt) {
        float pv = __expf(sacc[nt][r] * scale - nm);
        ls += pv;
        // C-format -> row-major P tile in wave-private LDS (in-order per wave)
        pw[(hi * 8 + r) * 64 + nt * 16 + ln] = (_Float16)pv;
      }
#pragma unroll
      for (int off = 8; off >= 1; off >>= 1) ls += __shfl_xor(ls, off, 16);
      runSum[r] = runSum[r] * alpha + ls;
#pragma unroll
      for (int dt = 0; dt < 4; ++dt) oa[dt][r] *= alpha;
    }

    // O += P V : reload P as A-fragments; V^T rows give contiguous B columns.
    {
      const _Float16* prow = pw + ln * 64;
      h16_t pa0 = load_a_frag(prow + 0, hi);
      h16_t pa1 = load_a_frag(prow + 32, hi);
#pragma unroll
      for (int dt = 0; dt < 4; ++dt) {
        const _Float16* vcol = VTc + (dt * 16 + ln) * 64;
        oa[dt] = wmma16x16x32(pa0, load_b_frag(vcol, hi), oa[dt]);
        oa[dt] = wmma16x16x32(pa1, load_b_frag(vcol + 32, hi), oa[dt]);
      }
    }
    __syncthreads();  // compute on buf[cur] done before it can be overwritten
    cur ^= 1;
  }

  // Normalize and scatter ctx back to [B,S,D_MODEL] f16 (head-interleaved).
#pragma unroll
  for (int r = 0; r < 8; ++r) {
    float inv = 1.0f / runSum[r];
    int row = qBase + w * 16 + hi * 8 + r;
    size_t orow = ((size_t)b * SEQ + row) * D_MODEL + h * HEAD_DIM;
#pragma unroll
    for (int dt = 0; dt < 4; ++dt)
      ctx[orow + dt * 16 + ln] = (_Float16)(oa[dt][r] * inv);
  }
}

// ---------------------------------------------------------------------------
// O-projection + bias + residual + LayerNorm, fused; pipelined WMMA GEMM.
// grid: NROWS/16   block: 256
// ---------------------------------------------------------------------------
__global__ void oproj_ln_kernel(const _Float16* __restrict__ ctx,
                                const _Float16* __restrict__ wo,
                                const float* __restrict__ bo,
                                const float* __restrict__ xres,
                                const float* __restrict__ gamma,
                                const float* __restrict__ beta,
                                float* __restrict__ out) {
  __shared__ float rowbuf[16 * D_MODEL];  // 32 KB
  const int tid = threadIdx.x;
  const int w = tid >> 5, lane = tid & 31, ln = lane & 15, hi = lane >> 4;
  const int mBase = blockIdx.x * 16;
  const int nBase = w * 64;

  const _Float16* arow = ctx + (size_t)(mBase + ln) * D_MODEL;
  const _Float16* brow = wo + (size_t)(nBase + ln) * D_MODEL;
  __builtin_prefetch(brow, 0, 0);

  v8f acc[4] = {};
  h16_t a_cur = load_a_frag(arow, hi);
  h16_t b_cur[4];
#pragma unroll
  for (int nt = 0; nt < 4; ++nt)
    b_cur[nt] = load_b_frag(brow + (size_t)nt * 16 * D_MODEL, hi);

  for (int kb = 0; kb < D_MODEL - 32; kb += 32) {
    h16_t a_nxt = load_a_frag(arow + kb + 32, hi);
    h16_t b_nxt[4];
#pragma unroll
    for (int nt = 0; nt < 4; ++nt)
      b_nxt[nt] = load_b_frag(brow + (size_t)nt * 16 * D_MODEL + kb + 32, hi);
#pragma unroll
    for (int nt = 0; nt < 4; ++nt)
      acc[nt] = wmma16x16x32(a_cur, b_cur[nt], acc[nt]);
    a_cur = a_nxt;
#pragma unroll
    for (int nt = 0; nt < 4; ++nt) b_cur[nt] = b_nxt[nt];
  }
#pragma unroll
  for (int nt = 0; nt < 4; ++nt)
    acc[nt] = wmma16x16x32(a_cur, b_cur[nt], acc[nt]);

#pragma unroll
  for (int nt = 0; nt < 4; ++nt) {
#pragma unroll
    for (int r = 0; r < 8; ++r) {
      int row = hi * 8 + r;
      int col = nBase + nt * 16 + ln;
      float v = acc[nt][r] + bo[col] +
                xres[(size_t)(mBase + row) * D_MODEL + col];
      rowbuf[row * D_MODEL + col] = v;
    }
  }
  __syncthreads();

  // LayerNorm: wave w handles rows 2w, 2w+1 (width-32 shuffle reductions).
#pragma unroll
  for (int rr = 0; rr < 2; ++rr) {
    int row = w * 2 + rr;
    float s = 0.f, ss = 0.f;
#pragma unroll
    for (int j = 0; j < 16; ++j) {
      float v = rowbuf[row * D_MODEL + lane + j * 32];
      s += v;
      ss += v * v;
    }
#pragma unroll
    for (int off = 16; off >= 1; off >>= 1) {
      s  += __shfl_xor(s, off, 32);
      ss += __shfl_xor(ss, off, 32);
    }
    float mean = s * (1.0f / D_MODEL);
    float var  = ss * (1.0f / D_MODEL) - mean * mean;
    float rstd = rsqrtf(var + 1e-5f);
    size_t orow = (size_t)(mBase + row) * D_MODEL;
#pragma unroll
    for (int j = 0; j < 16; ++j) {
      int col = lane + j * 32;
      out[orow + col] =
          (rowbuf[row * D_MODEL + col] - mean) * rstd * gamma[col] + beta[col];
    }
  }
}

// ---------------------------------------------------------------------------
// Host-side launcher
// ---------------------------------------------------------------------------
extern "C" void kernel_launch(void* const* d_in, const int* in_sizes, int n_in,
                              void* d_out, int out_size, void* d_ws,
                              size_t ws_size, hipStream_t stream) {
  (void)in_sizes; (void)n_in; (void)out_size; (void)ws_size;

  const float* x     = (const float*)d_in[0];
  const float* Wq    = (const float*)d_in[1];
  const float* bq    = (const float*)d_in[2];
  const float* Wk    = (const float*)d_in[3];
  const float* bk    = (const float*)d_in[4];
  const float* Wv    = (const float*)d_in[5];
  const float* bv    = (const float*)d_in[6];
  const float* Wo    = (const float*)d_in[7];
  const float* bo    = (const float*)d_in[8];
  const float* gamma = (const float*)d_in[9];
  const float* beta  = (const float*)d_in[10];
  float* out = (float*)d_out;

  // Workspace layout (bytes):
  //   xh @0 (16 MiB) | wh @16Mi (2 MiB, q/k/v/o) | qh | kh | vt | ctx (16 MiB each)
  char* ws = (char*)d_ws;
  const size_t SZ_X = (size_t)NROWS * D_MODEL * 2;
  const size_t SZ_W = (size_t)D_MODEL * D_MODEL * 2;
  _Float16* xh  = (_Float16*)(ws);
  _Float16* wh  = (_Float16*)(ws + SZ_X);
  _Float16* qh  = (_Float16*)(ws + SZ_X + 4 * SZ_W);
  _Float16* kh  = (_Float16*)(ws + SZ_X + 4 * SZ_W + SZ_X);
  _Float16* vt  = (_Float16*)(ws + SZ_X + 4 * SZ_W + 2 * SZ_X);
  _Float16* ctx = (_Float16*)(ws + SZ_X + 4 * SZ_W + 3 * SZ_X);

  const int n_x = NROWS * D_MODEL;         // 8,388,608
  const int n_w = D_MODEL * D_MODEL;       //   262,144
  cvt_f32_f16<<<n_x / 256, 256, 0, stream>>>(x, xh, n_x);
  cvt_f32_f16<<<n_w / 256, 256, 0, stream>>>(Wq, wh + 0 * (size_t)n_w, n_w);
  cvt_f32_f16<<<n_w / 256, 256, 0, stream>>>(Wk, wh + 1 * (size_t)n_w, n_w);
  cvt_f32_f16<<<n_w / 256, 256, 0, stream>>>(Wv, wh + 2 * (size_t)n_w, n_w);
  cvt_f32_f16<<<n_w / 256, 256, 0, stream>>>(Wo, wh + 3 * (size_t)n_w, n_w);

  qkv_proj_kernel<<<dim3(NROWS / 64, D_MODEL / 128, 3), 256, 0, stream>>>(
      xh, wh, bq, bk, bv, qh, kh, vt);

  flash_attn_kernel<<<dim3(SEQ / 64, BATCH * N_HEAD), 128, 0, stream>>>(
      qh, kh, vt, ctx);

  oproj_ln_kernel<<<NROWS / 16, 256, 0, stream>>>(
      ctx, wh + 3 * (size_t)n_w, bo, x, gamma, beta, out);
}
